// AttentionLayer_49727131353318
// MI455X (gfx1250) — matrix operand order repaired and checked by reading, not compiled, not run on hardware
//
#include <hip/hip_runtime.h>
#include <stdint.h>

#define B_ 4
#define C_ 256
#define N_ 4096
#define CR_ 32

typedef __bf16 bf16_t;
typedef __attribute__((ext_vector_type(16))) __bf16 v16bf;
typedef __attribute__((ext_vector_type(8)))  float  v8f;

union V16U { v16bf v; uint4 q[2]; };
union H8U  { __bf16 h[8]; uint4 q; };

// ---------- CDNA5 async global->LDS copy (ASYNCcnt-tracked) ----------
__device__ __forceinline__ void async_copy_b128(void* lds_dst, const void* gsrc) {
  uint32_t l = (uint32_t)(uintptr_t)lds_dst;   // flat->LDS uses addr[31:0]
  asm volatile("global_load_async_to_lds_b128 %0, %1, off"
               :: "v"(l), "v"(gsrc) : "memory");
}
__device__ __forceinline__ void wait_async0() {
  asm volatile("s_wait_asynccnt 0x0" ::: "memory");
}

__device__ __forceinline__ v8f wmma_bf16f32(v16bf a, v16bf b, v8f c) {
  return __builtin_amdgcn_wmma_f32_16x16x32_bf16(false, a, false, b, (short)0, c,
                                                 false, false);
}

// ---------------- kernel 1: convert weights to bf16 ----------------
__global__ __launch_bounds__(256) void prep_weights(
    const float* __restrict__ Wv, const float* __restrict__ Wk,
    const float* __restrict__ Wq, const float* __restrict__ Wo,
    bf16_t* __restrict__ wqkv, bf16_t* __restrict__ wo)
{
  int idx = blockIdx.x * 256 + threadIdx.x;           // 32768 total
  if (idx < 96 * 256) {
    int g = idx >> 8, cc = idx & 255;
    float v = (g < 32) ? Wv[g * 256 + cc]
            : (g < 64) ? Wk[(g - 32) * 256 + cc]
                       : Wq[(g - 64) * 256 + cc];
    wqkv[idx] = (bf16_t)v;
  } else {
    int j = idx - 96 * 256;
    if (j < 256 * 32) wo[j] = (bf16_t)Wo[j];
  }
}

// ---------------- kernel 2: fused QKV projection (WMMA) ----------------
// out[96, n] = Wqkv[96,256] @ x[b,:,n] + bias ; q,k stored transposed [N][32],
// v stored channel-major [32][N].
__global__ __launch_bounds__(256) void qkv_proj(
    const float* __restrict__ x, const bf16_t* __restrict__ wqkv,
    const float* __restrict__ bv, const float* __restrict__ bk,
    const float* __restrict__ bq,
    bf16_t* __restrict__ qT, bf16_t* __restrict__ kT, bf16_t* __restrict__ vC)
{
  int tid = threadIdx.x, lane = tid & 31, wave = tid >> 5;
  int lm = lane & 15, ph = lane >> 4;                 // lane half: 0/1
  int b = blockIdx.x >> 5;                            // 32 n-blocks per batch
  int nblk = blockIdx.x & 31;
  int n = nblk * 128 + wave * 16 + lm;                // this lane's column

  v8f acc[6] = {};
  #pragma unroll
  for (int k = 0; k < 8; ++k) {
    int k0 = k * 32;
    // B-tile: x[b, k0..k0+31, n] ; lane holds 16 contiguous K values
    v16bf xb;
    const float* xp = x + ((size_t)(b * C_ + k0 + ph * 16)) * N_ + n;
    #pragma unroll
    for (int t = 0; t < 16; ++t) xb[t] = (bf16_t)xp[(size_t)t * N_];
    #pragma unroll
    for (int rt = 0; rt < 6; ++rt) {
      // A-tile: W rows rt*16..+15, K window k0..k0+31 (16 contiguous K per lane half)
      const char* wr = (const char*)wqkv +
                       ((size_t)(rt * 16 + lm) * 256 + k0) * 2 + ph * 32;
      V16U wa;
      wa.q[0] = *(const uint4*)(wr);
      wa.q[1] = *(const uint4*)(wr + 16);
      acc[rt] = wmma_bf16f32(wa.v, xb, acc[rt]);
    }
  }
  #pragma unroll
  for (int rt = 0; rt < 6; ++rt) {
    int cb = ((rt & 1) << 4) + (ph << 3);             // channel base within proj
    int p = rt >> 1;                                  // 0=f(Q) 1=g(K) 2=h(V)
    const float* bp = (p == 0) ? bv : (p == 1) ? bk : bq;
    if (rt < 4) {
      bf16_t* dst = (p == 0) ? qT : kT;
      H8U h;
      #pragma unroll
      for (int r = 0; r < 8; ++r) h.h[r] = (bf16_t)(acc[rt][r] + bp[cb + r]);
      *(uint4*)(dst + ((size_t)(b * N_ + n)) * CR_ + cb) = h.q;
    } else {
      #pragma unroll
      for (int r = 0; r < 8; ++r)
        vC[((size_t)(b * CR_ + cb + r)) * N_ + n] = (bf16_t)(acc[rt][r] + bp[cb + r]);
    }
  }
}

// ---------------- kernel 3: flash attention (double-buffered async) ----------
// block = 8 waves, 128 query rows; key blocks of 128, K/V prefetched into the
// alternate LDS buffer while the current one is consumed (ASYNCcnt in-order).
__global__ __launch_bounds__(256) void flash_attn(
    const bf16_t* __restrict__ qT, const bf16_t* __restrict__ kT,
    const bf16_t* __restrict__ vC, bf16_t* __restrict__ outT)
{
  __shared__ uint4 ldsK4[2][128 * 4];   // kT tile [128 j][32 c] bf16 = 8KB x2
  __shared__ uint4 ldsV4[2][32 * 16];   // vC tile [32 c][128 j] bf16 = 8KB x2

  int tid = threadIdx.x, lane = tid & 31, wave = tid >> 5;
  int lm = lane & 15, ph = lane >> 4;
  int b = blockIdx.x >> 5;
  int iblk = blockIdx.x & 31;
  int iBase = iblk * 128 + wave * 16;

  // Q B-tile (K=c contiguous per lane), held all kernel
  const char* qrow = (const char*)qT +
                     ((size_t)(b * N_ + iBase + lm)) * CR_ * 2 + ph * 32;
  V16U qb;
  qb.q[0] = *(const uint4*)qrow;
  qb.q[1] = *(const uint4*)(qrow + 16);

  auto stage = [&](int buf, int j0) {
    // K tile: 128 consecutive rows of kT -> one contiguous 8KB region
    const char* ks = (const char*)kT + ((size_t)(b * N_ + j0)) * CR_ * 2;
    async_copy_b128((char*)ldsK4[buf] + tid * 16, ks + tid * 16);
    async_copy_b128((char*)ldsK4[buf] + (tid + 256) * 16,
                    ks + (size_t)(tid + 256) * 16);
    // V tile: 32 rows of 256B, row stride N_*2
    #pragma unroll
    for (int cpy = 0; cpy < 2; ++cpy) {
      int idx = tid + cpy * 256;
      int row = idx >> 4, col = idx & 15;
      const char* vs = (const char*)vC +
                       ((size_t)(b * CR_ + row) * N_ + j0) * 2 + col * 16;
      async_copy_b128((char*)ldsV4[buf] + idx * 16, vs);
    }
  };

  v8f o0 = {}, o1 = {};
  float m = -3.0e38f, l = 0.f;

  stage(0, 0);                                        // prime the pipeline
  for (int it = 0; it < N_ / 128; ++it) {
    int cur = it & 1;
    wait_async0();       // only tile `it` is in flight here (in-order ASYNCcnt)
    __syncthreads();     // everyone's copies landed; iter it-1 compute done
    if (it + 1 < N_ / 128) stage(1 - cur, (it + 1) * 128);  // prefetch next

    const uint4* K4 = ldsK4[cur];
    const uint4* V4 = ldsV4[cur];

    // S^T tiles: D[j,i] = Ktile(16j x 32c) x Q(32c x 16i)
    v8f s[8];
    #pragma unroll
    for (int t = 0; t < 8; ++t) {
      int jl = t * 16 + lm;
      V16U ka;
      ka.q[0] = K4[jl * 4 + ph * 2];
      ka.q[1] = K4[jl * 4 + ph * 2 + 1];
      v8f z = {};
      s[t] = wmma_bf16f32(ka.v, qb.v, z);
    }

    // online softmax: lane holds query i=lm (both halves), disjoint j sets
    float mloc = -3.0e38f;
    #pragma unroll
    for (int t = 0; t < 8; ++t)
      #pragma unroll
      for (int r = 0; r < 8; ++r) mloc = fmaxf(mloc, s[t][r]);
    mloc = fmaxf(mloc, __shfl_xor(mloc, 16));
    float mnew = fmaxf(m, mloc);
    float alpha = __expf(m - mnew);
    m = mnew;
    float ls = 0.f;
    #pragma unroll
    for (int t = 0; t < 8; ++t)
      #pragma unroll
      for (int r = 0; r < 8; ++r) {
        float p = __expf(s[t][r] - mnew);
        s[t][r] = p;
        ls += p;
      }
    ls += __shfl_xor(ls, 16);
    l = l * alpha + ls;

    // rescale output accumulators (alpha per row, broadcast via shuffle)
    #pragma unroll
    for (int r = 0; r < 8; ++r) {
      float ar = __shfl(alpha, r + ((lane & 16) >> 1));
      o0[r] *= ar;
      o1[r] *= ar;
    }

    // P·V : S^T D-layout packs directly into PV A-layout
    #pragma unroll
    for (int a = 0; a < 4; ++a) {
      v16bf pa;
      #pragma unroll
      for (int r = 0; r < 8; ++r) {
        pa[r]     = (bf16_t)s[2 * a][r];
        pa[8 + r] = (bf16_t)s[2 * a + 1][r];
      }
      V16U vb0, vb1;
      int i0v = lm * 16 + a * 4 + ph;
      vb0.q[0] = V4[i0v];
      vb0.q[1] = V4[i0v + 2];
      int i1v = (16 + lm) * 16 + a * 4 + ph;
      vb1.q[0] = V4[i1v];
      vb1.q[1] = V4[i1v + 2];
      o0 = wmma_bf16f32(pa, vb0.v, o0);
      o1 = wmma_bf16f32(pa, vb1.v, o1);
    }
  }

  float linv = 1.0f / l;
  #pragma unroll
  for (int r = 0; r < 8; ++r) {
    float sr = __shfl(linv, r + ((lane & 16) >> 1));
    int i = iBase + r + ((lane & 16) >> 1);
    bf16_t* d0 = outT + ((size_t)(b * N_ + i)) * CR_ + lm;
    d0[0]  = (bf16_t)(o0[r] * sr);
    d0[16] = (bf16_t)(o1[r] * sr);
  }
}

// ---------------- kernel 4: output projection + residual ----------------
__global__ __launch_bounds__(256) void out_proj(
    const bf16_t* __restrict__ oT, const bf16_t* __restrict__ wo,
    const float* __restrict__ bo, const float* __restrict__ gamma,
    const float* __restrict__ x, float* __restrict__ out)
{
  int tid = threadIdx.x, lane = tid & 31, wave = tid >> 5;
  int lm = lane & 15, ph = lane >> 4;
  int tile = blockIdx.x * 8 + wave;                  // 16384 tiles
  int b  = tile >> 12;
  int it = (tile >> 4) & 255;
  int ct = tile & 15;

  const char* ar = (const char*)oT +
                   ((size_t)(b * N_ + it * 16 + lm)) * CR_ * 2 + ph * 32;
  V16U A;
  A.q[0] = *(const uint4*)ar;
  A.q[1] = *(const uint4*)(ar + 16);
  const char* br = (const char*)wo + ((size_t)(ct * 16 + lm)) * CR_ * 2 + ph * 32;
  V16U Bm;
  Bm.q[0] = *(const uint4*)br;
  Bm.q[1] = *(const uint4*)(br + 16);
  v8f z = {};
  v8f d = wmma_bf16f32(A.v, Bm.v, z);                // D[i, c_out], K = CR = 32

  int c  = ct * 16 + lm;
  int i0 = it * 16 + (ph << 3);
  float g  = gamma[0];
  float bc = bo[c];
  const float* xr = x + ((size_t)(b * C_ + c)) * N_ + i0;
  float* orow     = out + ((size_t)(b * C_ + c)) * N_ + i0;
  #pragma unroll
  for (int r = 0; r < 8; ++r) orow[r] = g * (d[r] + bc) + xr[r];
}

// ---------------- launcher ----------------
extern "C" void kernel_launch(void* const* d_in, const int* in_sizes, int n_in,
                              void* d_out, int out_size, void* d_ws, size_t ws_size,
                              hipStream_t stream) {
  const float* x     = (const float*)d_in[0];
  const float* Wv    = (const float*)d_in[1];
  const float* bv    = (const float*)d_in[2];
  const float* Wk    = (const float*)d_in[3];
  const float* bk    = (const float*)d_in[4];
  const float* Wq    = (const float*)d_in[5];
  const float* bq    = (const float*)d_in[6];
  const float* Wo    = (const float*)d_in[7];
  const float* bo    = (const float*)d_in[8];
  const float* gamma = (const float*)d_in[9];

  char* ws = (char*)d_ws;
  const size_t T = (size_t)B_ * N_ * CR_ * 2;        // 1 MB per tensor
  bf16_t* qT   = (bf16_t*)(ws);
  bf16_t* kT   = (bf16_t*)(ws + T);
  bf16_t* vC   = (bf16_t*)(ws + 2 * T);
  bf16_t* oT   = (bf16_t*)(ws + 3 * T);
  bf16_t* wqkv = (bf16_t*)(ws + 4 * T);
  bf16_t* wo   = (bf16_t*)(ws + 4 * T + (size_t)96 * 256 * 2);

  prep_weights<<<128, 256, 0, stream>>>(Wv, Wk, Wq, Wo, wqkv, wo);
  qkv_proj<<<128, 256, 0, stream>>>(x, wqkv, bv, bk, bq, qT, kT, vC);
  flash_attn<<<B_ * (N_ / 128), 256, 0, stream>>>(qT, kT, vC, oT);
  out_proj<<<2048, 256, 0, stream>>>(oT, wo, bo, gamma, x, (float*)d_out);
}
